// GNN_45045617001165
// MI455X (gfx1250) — matrix operand order repaired and checked by reading, not compiled, not run on hardware
//
#include <hip/hip_runtime.h>
#include <hip/hip_fp16.h>

typedef __attribute__((ext_vector_type(16))) _Float16 v16h;
typedef __attribute__((ext_vector_type(8)))  _Float16 v8h;
typedef __attribute__((ext_vector_type(8)))  float    v8f;
typedef __attribute__((ext_vector_type(4)))  float    v4f;

#define DEV __device__ __forceinline__

DEV float silu_f(float x) { return x / (1.0f + __expf(-x)); }
DEV bool uniform(bool c) { return __builtin_amdgcn_readfirstlane((int)c) != 0; }

// ---- CDNA5 WMMA fragment builders (ISA 7.12.2 layouts, wave32) ----
// A 16x32 f16: lane m = lane&15 holds row m. sel = lane>>4.
//   elems 0..7  = K (8*sel + 0..7), elems 8..15 = K (16 + 8*sel + 0..7)
DEV v16h a_from_f32(const float* __restrict__ src, int sel) {
  v4f f0 = *(const v4f*)(src + 8 * sel);
  v4f f1 = *(const v4f*)(src + 8 * sel + 4);
  v4f f2 = *(const v4f*)(src + 16 + 8 * sel);
  v4f f3 = *(const v4f*)(src + 16 + 8 * sel + 4);
  v16h a;
#pragma unroll
  for (int i = 0; i < 4; ++i) {
    a[i]      = (_Float16)f0[i];
    a[4 + i]  = (_Float16)f1[i];
    a[8 + i]  = (_Float16)f2[i];
    a[12 + i] = (_Float16)f3[i];
  }
  return a;
}

// Same but only local K 0..15 valid (K 16..31 are zero padding).
DEV v16h a_from_f32_half(const float* __restrict__ src, int sel) {
  v4f f0 = *(const v4f*)(src + 8 * sel);
  v4f f1 = *(const v4f*)(src + 8 * sel + 4);
  v16h a;
#pragma unroll
  for (int i = 0; i < 4; ++i) {
    a[i]       = (_Float16)f0[i];
    a[4 + i]   = (_Float16)f1[i];
    a[8 + i]   = (_Float16)0.0f;
    a[12 + i]  = (_Float16)0.0f;
  }
  return a;
}

// A fragment from an LDS row of f16 (row = 64 contiguous halfs for this M).
DEV v16h a_from_lds(const _Float16* row, int kbase, int sel) {
  v8h lo = *(const v8h*)(row + kbase + 8 * sel);
  v8h hi = *(const v8h*)(row + kbase + 16 + 8 * sel);
  v16h a;
#pragma unroll
  for (int i = 0; i < 8; ++i) { a[i] = lo[i]; a[8 + i] = hi[i]; }
  return a;
}

// Single-tile K-step.
template <int NTC>
DEV void wmma_k(v8f* acc, v16h a, const _Float16* __restrict__ wf, int ks, int lane) {
#pragma unroll
  for (int nt = 0; nt < NTC; ++nt) {
    v16h b = *(const v16h*)(wf + (size_t)(((ks * NTC + nt) * 32 + lane) * 16));
    acc[nt] = __builtin_amdgcn_wmma_f32_16x16x32_f16(
        false, a, false, b, (short)0, acc[nt], false, false);
  }
}

// Dual-tile K-step: one B-fragment load feeds two independent WMMAs.
template <int NTC>
DEV void wmma_k2(v8f* accA, v8f* accB, v16h aA, v16h aB,
                 const _Float16* __restrict__ wf, int ks, int lane) {
#pragma unroll
  for (int nt = 0; nt < NTC; ++nt) {
    v16h b = *(const v16h*)(wf + (size_t)(((ks * NTC + nt) * 32 + lane) * 16));
    accA[nt] = __builtin_amdgcn_wmma_f32_16x16x32_f16(
        false, aA, false, b, (short)0, accA[nt], false, false);
    accB[nt] = __builtin_amdgcn_wmma_f32_16x16x32_f16(
        false, aB, false, b, (short)0, accB[nt], false, false);
  }
}

#define LDS_FENCE() asm volatile("s_wait_dscnt 0x0" ::: "memory")

// ---- Weight repack: W[Klog x Nout] f32 row-major -> f16 B fragments ----
// B 32x16 f16 layout: column N = lane&15; elems 0..7 = K (8*sel + j), elems 8..15 = K (16 + 8*sel + j-8).
__global__ void gnn_repack_w(const float* __restrict__ W, _Float16* __restrict__ dst,
                             int Klog, int ksc, int ntc, int Nout) {
  int t = blockIdx.x * blockDim.x + threadIdx.x;
  int total = ksc * ntc * 32;
  if (t >= total) return;
  int lane = t & 31, rest = t >> 5;
  int nt = rest % ntc, ks = rest / ntc;
  int sel = lane >> 4, n = lane & 15;
  _Float16* o = dst + (size_t)((ks * ntc + nt) * 32 + lane) * 16;
#pragma unroll
  for (int j = 0; j < 16; ++j) {
    int k = ks * 32 + (j < 8 ? 8 * sel + j : 16 + 8 * sel + (j - 8));
    float v = (k < Klog) ? W[(size_t)k * Nout + nt * 16 + n] : 0.0f;
    o[j] = (_Float16)v;
  }
}

// ---- Embedding: h = nf @ emb_W + emb_b  ([N,16] @ [16,64]) ----
__global__ __launch_bounds__(256) void gnn_embed(
    const float* __restrict__ nf, const _Float16* __restrict__ wf,
    const float* __restrict__ b, float* __restrict__ h, int N) {
  __shared__ float ls[8][16 * 64];
  const int wave = threadIdx.x >> 5, lane = threadIdx.x & 31;
  const int sel = lane >> 4, ln = lane & 15;
  const int ntiles = (N + 15) >> 4;
  const int tile = blockIdx.x * 8 + wave;
  if (tile >= ntiles) return;
  const int n0 = tile << 4;
  int nm = n0 + ln; if (nm >= N) nm = N - 1;
  v8f zero = {};
  v8f acc[4];
#pragma unroll
  for (int nt = 0; nt < 4; ++nt) acc[nt] = zero;
  wmma_k<4>(acc, a_from_f32_half(nf + (size_t)nm * 16, sel), wf, 0, lane);

  if (uniform(n0 + 16 <= N)) {
    float* mf = ls[wave];
#pragma unroll
    for (int nt = 0; nt < 4; ++nt) {
      float bb = b[nt * 16 + ln];
#pragma unroll
      for (int rr = 0; rr < 8; ++rr)
        mf[(rr + 8 * sel) * 64 + nt * 16 + ln] = acc[nt][rr] + bb;
    }
    LDS_FENCE();
    float* dst = h + (size_t)n0 * 64;
#pragma unroll
    for (int it = 0; it < 8; ++it) {
      int o = it * 128 + lane * 4;
      *(v4f*)(dst + o) = *(const v4f*)(mf + o);
    }
  } else {
#pragma unroll
    for (int nt = 0; nt < 4; ++nt) {
      float bb = b[nt * 16 + ln];
#pragma unroll
      for (int rr = 0; rr < 8; ++rr) {
        int node = n0 + rr + 8 * sel;
        if (node < N) h[(size_t)node * 64 + nt * 16 + ln] = acc[nt][rr] + bb;
      }
    }
  }
}

// ---- Fused edge pipeline (2 tiles / wave): gather -> MLP1 -> SiLU -> MLP2 -> SiLU -> scatter ----
__global__ __launch_bounds__(256) void gnn_edge(
    const float* __restrict__ h, const int* __restrict__ erow, const int* __restrict__ ecol,
    const float* __restrict__ ef,
    const _Float16* __restrict__ w1, const float* __restrict__ b1,
    const _Float16* __restrict__ w2, const float* __restrict__ b2,
    float* __restrict__ agg, int E) {
  __shared__ _Float16 lm[8][2][16 * 64];
  __shared__ int lrow[8][32];
  const int wave = threadIdx.x >> 5, lane = threadIdx.x & 31;
  const int sel = lane >> 4, ln = lane & 15;
  const int ntiles = (E + 15) >> 4;
  const int npairs = (ntiles + 1) >> 1;
  const int p = blockIdx.x * 8 + wave;
  if (p >= npairs) return;
  const int e0 = p << 5;  // 32 edges per wave

  { int e = e0 + lane; if (e >= E) e = E - 1; lrow[wave][lane] = erow[e]; }
  LDS_FENCE();

  int eA = e0 + ln;       if (eA >= E) eA = E - 1;
  int eB = e0 + 16 + ln;  if (eB >= E) eB = E - 1;
  const int rA = lrow[wave][ln], rB = lrow[wave][16 + ln];
  const int cA = ecol[eA], cB = ecol[eB];
  const float* hrA = h + (size_t)rA * 64;
  const float* hrB = h + (size_t)rB * 64;
  const float* hcA = h + (size_t)cA * 64;
  const float* hcB = h + (size_t)cB * 64;

  v8f zero = {};
  v8f accA[4], accB[4];
#pragma unroll
  for (int nt = 0; nt < 4; ++nt) { accA[nt] = zero; accB[nt] = zero; }

  // K = [h_row(64) | h_col(64) | ef(16) pad 16] = 160 -> 5 k-steps of 32
  wmma_k2<4>(accA, accB, a_from_f32(hrA,      sel), a_from_f32(hrB,      sel), w1, 0, lane);
  wmma_k2<4>(accA, accB, a_from_f32(hrA + 32, sel), a_from_f32(hrB + 32, sel), w1, 1, lane);
  wmma_k2<4>(accA, accB, a_from_f32(hcA,      sel), a_from_f32(hcB,      sel), w1, 2, lane);
  wmma_k2<4>(accA, accB, a_from_f32(hcA + 32, sel), a_from_f32(hcB + 32, sel), w1, 3, lane);
  wmma_k2<4>(accA, accB, a_from_f32_half(ef + (size_t)eA * 16, sel),
                          a_from_f32_half(ef + (size_t)eB * 16, sel), w1, 4, lane);

  // bias + SiLU, D layout (M = rr + 8*sel, col = nt*16 + ln) -> LDS f16
  _Float16* m0 = lm[wave][0];
  _Float16* m1 = lm[wave][1];
#pragma unroll
  for (int nt = 0; nt < 4; ++nt) {
    float bb = b1[nt * 16 + ln];
#pragma unroll
    for (int rr = 0; rr < 8; ++rr) {
      int o = (rr + 8 * sel) * 64 + nt * 16 + ln;
      m0[o] = (_Float16)silu_f(accA[nt][rr] + bb);
      m1[o] = (_Float16)silu_f(accB[nt][rr] + bb);
    }
  }
  LDS_FENCE();

  const _Float16* mr0 = m0 + ln * 64;
  const _Float16* mr1 = m1 + ln * 64;
  v8f acc2A[4], acc2B[4];
#pragma unroll
  for (int nt = 0; nt < 4; ++nt) { acc2A[nt] = zero; acc2B[nt] = zero; }
  wmma_k2<4>(acc2A, acc2B, a_from_lds(mr0, 0,  sel), a_from_lds(mr1, 0,  sel), w2, 0, lane);
  wmma_k2<4>(acc2A, acc2B, a_from_lds(mr0, 32, sel), a_from_lds(mr1, 32, sel), w2, 1, lane);

  // bias + SiLU + scatter-add to receiver nodes
  if (uniform(e0 + 32 <= E)) {
#pragma unroll
    for (int nt = 0; nt < 4; ++nt) {
      float bb = b2[nt * 16 + ln];
      int col = nt * 16 + ln;
#pragma unroll
      for (int rr = 0; rr < 8; ++rr) {
        int M = rr + 8 * sel;
        atomicAdd(&agg[(size_t)lrow[wave][M] * 64 + col],      silu_f(acc2A[nt][rr] + bb));
        atomicAdd(&agg[(size_t)lrow[wave][16 + M] * 64 + col], silu_f(acc2B[nt][rr] + bb));
      }
    }
  } else {
#pragma unroll
    for (int nt = 0; nt < 4; ++nt) {
      float bb = b2[nt * 16 + ln];
      int col = nt * 16 + ln;
#pragma unroll
      for (int rr = 0; rr < 8; ++rr) {
        int M = rr + 8 * sel;
        if (e0 + M < E)
          atomicAdd(&agg[(size_t)lrow[wave][M] * 64 + col], silu_f(acc2A[nt][rr] + bb));
        if (e0 + 16 + M < E)
          atomicAdd(&agg[(size_t)lrow[wave][16 + M] * 64 + col], silu_f(acc2B[nt][rr] + bb));
      }
    }
  }
}

// ---- Node residual MLP (2 tiles / wave): h' = h + relu([h|agg]@Wn1+bn1)@Wn2 + bn2 ----
__global__ __launch_bounds__(256) void gnn_node(
    const float* __restrict__ h_in, const float* __restrict__ agg,
    const _Float16* __restrict__ w1, const float* __restrict__ b1,
    const _Float16* __restrict__ w2, const float* __restrict__ b2,
    float* __restrict__ h_out, int N) {
  __shared__ _Float16 lm[8][2][16 * 64];
  const int wave = threadIdx.x >> 5, lane = threadIdx.x & 31;
  const int sel = lane >> 4, ln = lane & 15;
  const int ntiles = (N + 15) >> 4;
  const int npairs = (ntiles + 1) >> 1;
  const int p = blockIdx.x * 8 + wave;
  if (p >= npairs) return;
  const int n0 = p << 5;  // 32 nodes per wave
  int nmA = n0 + ln;      if (nmA >= N) nmA = N - 1;
  int nmB = n0 + 16 + ln; if (nmB >= N) nmB = N - 1;
  const float* hrA = h_in + (size_t)nmA * 64;
  const float* hrB = h_in + (size_t)nmB * 64;
  const float* arA = agg + (size_t)nmA * 64;
  const float* arB = agg + (size_t)nmB * 64;

  v8f zero = {};
  v8f accA[4], accB[4];
#pragma unroll
  for (int nt = 0; nt < 4; ++nt) { accA[nt] = zero; accB[nt] = zero; }
  wmma_k2<4>(accA, accB, a_from_f32(hrA,      sel), a_from_f32(hrB,      sel), w1, 0, lane);
  wmma_k2<4>(accA, accB, a_from_f32(hrA + 32, sel), a_from_f32(hrB + 32, sel), w1, 1, lane);
  wmma_k2<4>(accA, accB, a_from_f32(arA,      sel), a_from_f32(arB,      sel), w1, 2, lane);
  wmma_k2<4>(accA, accB, a_from_f32(arA + 32, sel), a_from_f32(arB + 32, sel), w1, 3, lane);

  _Float16* m0 = lm[wave][0];
  _Float16* m1 = lm[wave][1];
#pragma unroll
  for (int nt = 0; nt < 4; ++nt) {
    float bb = b1[nt * 16 + ln];
#pragma unroll
    for (int rr = 0; rr < 8; ++rr) {
      int o = (rr + 8 * sel) * 64 + nt * 16 + ln;
      float vA = accA[nt][rr] + bb; vA = vA > 0.0f ? vA : 0.0f;
      float vB = accB[nt][rr] + bb; vB = vB > 0.0f ? vB : 0.0f;
      m0[o] = (_Float16)vA;
      m1[o] = (_Float16)vB;
    }
  }
  LDS_FENCE();

  const _Float16* mr0 = m0 + ln * 64;
  const _Float16* mr1 = m1 + ln * 64;
  v8f acc2A[4], acc2B[4];
#pragma unroll
  for (int nt = 0; nt < 4; ++nt) { acc2A[nt] = zero; acc2B[nt] = zero; }
  wmma_k2<4>(acc2A, acc2B, a_from_lds(mr0, 0,  sel), a_from_lds(mr1, 0,  sel), w2, 0, lane);
  wmma_k2<4>(acc2A, acc2B, a_from_lds(mr0, 32, sel), a_from_lds(mr1, 32, sel), w2, 1, lane);

  float bbv[4];
#pragma unroll
  for (int nt = 0; nt < 4; ++nt) bbv[nt] = b2[nt * 16 + ln];

  if (uniform(n0 + 32 <= N)) {
    // stage tile A as f32 in the (now free) LDS region; coalesced vector epilogue
    float* mf = (float*)lm[wave];
    LDS_FENCE();  // f16 reads done before overwrite
#pragma unroll
    for (int nt = 0; nt < 4; ++nt)
#pragma unroll
      for (int rr = 0; rr < 8; ++rr)
        mf[(rr + 8 * sel) * 64 + nt * 16 + ln] = acc2A[nt][rr] + bbv[nt];
    LDS_FENCE();
    const float* hsrcA = h_in + (size_t)n0 * 64;
    float* hdstA = h_out + (size_t)n0 * 64;
#pragma unroll
    for (int it = 0; it < 8; ++it) {
      int o = it * 128 + lane * 4;
      v4f v = *(const v4f*)(mf + o);
      v4f rres = *(const v4f*)(hsrcA + o);
      *(v4f*)(hdstA + o) = v + rres;
    }
    LDS_FENCE();
    // tile B reuses the same staging region
#pragma unroll
    for (int nt = 0; nt < 4; ++nt)
#pragma unroll
      for (int rr = 0; rr < 8; ++rr)
        mf[(rr + 8 * sel) * 64 + nt * 16 + ln] = acc2B[nt][rr] + bbv[nt];
    LDS_FENCE();
    const float* hsrcB = h_in + (size_t)(n0 + 16) * 64;
    float* hdstB = h_out + (size_t)(n0 + 16) * 64;
#pragma unroll
    for (int it = 0; it < 8; ++it) {
      int o = it * 128 + lane * 4;
      v4f v = *(const v4f*)(mf + o);
      v4f rres = *(const v4f*)(hsrcB + o);
      *(v4f*)(hdstB + o) = v + rres;
    }
  } else {
#pragma unroll
    for (int nt = 0; nt < 4; ++nt) {
      int f = nt * 16 + ln;
#pragma unroll
      for (int rr = 0; rr < 8; ++rr) {
        int nodeA = n0 + rr + 8 * sel;
        int nodeB = n0 + 16 + rr + 8 * sel;
        if (nodeA < N)
          h_out[(size_t)nodeA * 64 + f] = acc2A[nt][rr] + bbv[nt] + h_in[(size_t)nodeA * 64 + f];
        if (nodeB < N)
          h_out[(size_t)nodeB * 64 + f] = acc2B[nt][rr] + bbv[nt] + h_in[(size_t)nodeB * 64 + f];
      }
    }
  }
}

// ---- Unembedding: out = silu(h@Wu1+bu1)@Wu2 + bu2 ----
__global__ __launch_bounds__(256) void gnn_unembed(
    const float* __restrict__ h,
    const _Float16* __restrict__ w1, const float* __restrict__ b1,
    const _Float16* __restrict__ w2, const float* __restrict__ b2,
    float* __restrict__ out, int N) {
  __shared__ float ls[8][16 * 64];
  const int wave = threadIdx.x >> 5, lane = threadIdx.x & 31;
  const int sel = lane >> 4, ln = lane & 15;
  const int ntiles = (N + 15) >> 4;
  const int tile = blockIdx.x * 8 + wave;
  if (tile >= ntiles) return;
  const int n0 = tile << 4;
  int nm = n0 + ln; if (nm >= N) nm = N - 1;
  const float* hr = h + (size_t)nm * 64;

  v8f zero = {};
  v8f acc[4];
#pragma unroll
  for (int nt = 0; nt < 4; ++nt) acc[nt] = zero;
  wmma_k<4>(acc, a_from_f32(hr,      sel), w1, 0, lane);
  wmma_k<4>(acc, a_from_f32(hr + 32, sel), w1, 1, lane);

  _Float16* m = (_Float16*)ls[wave];
#pragma unroll
  for (int nt = 0; nt < 4; ++nt) {
    float bb = b1[nt * 16 + ln];
#pragma unroll
    for (int rr = 0; rr < 8; ++rr) {
      float v = silu_f(acc[nt][rr] + bb);
      m[(rr + 8 * sel) * 64 + nt * 16 + ln] = (_Float16)v;
    }
  }
  LDS_FENCE();

  const _Float16* mr = m + ln * 64;
  v8f acc2[1] = {zero};
  wmma_k<1>(acc2, a_from_lds(mr, 0,  sel), w2, 0, lane);
  wmma_k<1>(acc2, a_from_lds(mr, 32, sel), w2, 1, lane);

  float bb = b2[ln];
  if (uniform(n0 + 16 <= N)) {
    float* mf = ls[wave];
    LDS_FENCE();
#pragma unroll
    for (int rr = 0; rr < 8; ++rr)
      mf[(rr + 8 * sel) * 16 + ln] = acc2[0][rr] + bb;
    LDS_FENCE();
    float* dst = out + (size_t)n0 * 16;
#pragma unroll
    for (int it = 0; it < 2; ++it) {
      int o = it * 128 + lane * 4;
      *(v4f*)(dst + o) = *(const v4f*)(mf + o);
    }
  } else {
#pragma unroll
    for (int rr = 0; rr < 8; ++rr) {
      int node = n0 + rr + 8 * sel;
      if (node < N) out[(size_t)node * 16 + ln] = acc2[0][rr] + bb;
    }
  }
}

extern "C" void kernel_launch(void* const* d_in, const int* in_sizes, int n_in,
                              void* d_out, int out_size, void* d_ws, size_t ws_size,
                              hipStream_t stream) {
  (void)n_in; (void)out_size; (void)ws_size;
  const float* node_features  = (const float*)d_in[0];
  const int*   edge_indices   = (const int*)d_in[1];
  const float* edges_features = (const float*)d_in[2];
  const float* emb_W = (const float*)d_in[4];
  const float* emb_b = (const float*)d_in[5];
  const float* We1 = (const float*)d_in[6];
  const float* be1 = (const float*)d_in[7];
  const float* We2 = (const float*)d_in[8];
  const float* be2 = (const float*)d_in[9];
  const float* Wn1 = (const float*)d_in[10];
  const float* bn1 = (const float*)d_in[11];
  const float* Wn2 = (const float*)d_in[12];
  const float* bn2 = (const float*)d_in[13];
  const float* Wu1 = (const float*)d_in[14];
  const float* bu1 = (const float*)d_in[15];
  const float* Wu2 = (const float*)d_in[16];
  const float* bu2 = (const float*)d_in[17];

  const int N = in_sizes[0] / 16;   // input_node_dim = 16
  const int E = in_sizes[1] / 2;    // edge_indices [2, E]
  const int H = 64, L = 4;

  // workspace carve-out (256B aligned)
  char* wsb = (char*)d_ws;
  size_t off = 0;
  auto alloc = [&](size_t bytes) -> char* {
    off = (off + 255) & ~(size_t)255;
    char* p = wsb + off; off += bytes; return p;
  };
  float* hA  = (float*)alloc((size_t)N * H * sizeof(float));
  float* hB  = (float*)alloc((size_t)N * H * sizeof(float));
  float* agg = (float*)alloc((size_t)N * H * sizeof(float));
  auto frag = [&](int ksc, int ntc) -> _Float16* {
    return (_Float16*)alloc((size_t)ksc * ntc * 512 * sizeof(_Float16));
  };
  _Float16* embF = frag(1, 4);
  _Float16 *We1F[4], *We2F[4], *Wn1F[4], *Wn2F[4];
  for (int l = 0; l < L; ++l) {
    We1F[l] = frag(5, 4);
    We2F[l] = frag(2, 4);
    Wn1F[l] = frag(4, 4);
    Wn2F[l] = frag(2, 4);
  }
  _Float16* Wu1F = frag(2, 4);
  _Float16* Wu2F = frag(2, 1);

  auto repack = [&](const float* W, _Float16* dst, int Klog, int ksc, int ntc, int Nout) {
    int total = ksc * ntc * 32;
    gnn_repack_w<<<(total + 255) / 256, 256, 0, stream>>>(W, dst, Klog, ksc, ntc, Nout);
  };
  repack(emb_W, embF, 16, 1, 4, 64);
  for (int l = 0; l < L; ++l) {
    repack(We1 + (size_t)l * 144 * 64, We1F[l], 144, 5, 4, 64);
    repack(We2 + (size_t)l * 64 * 64,  We2F[l], 64,  2, 4, 64);
    repack(Wn1 + (size_t)l * 128 * 64, Wn1F[l], 128, 4, 4, 64);
    repack(Wn2 + (size_t)l * 64 * 64,  Wn2F[l], 64,  2, 4, 64);
  }
  repack(Wu1, Wu1F, 64, 2, 4, 64);
  repack(Wu2, Wu2F, 64, 2, 1, 16);

  const int ntilesN = (N + 15) / 16;
  const int ntilesE = (E + 15) / 16;
  const int nblkN = (ntilesN + 7) / 8;                       // embed / unembed
  const int npairsE = (ntilesE + 1) / 2, nblkE = (npairsE + 7) / 8;
  const int npairsN = (ntilesN + 1) / 2, nblkNode = (npairsN + 7) / 8;
  const int* erow = edge_indices;
  const int* ecol = edge_indices + E;

  gnn_embed<<<nblkN, 256, 0, stream>>>(node_features, embF, emb_b, hA, N);

  float* hc = hA; float* hn = hB;
  for (int l = 0; l < L; ++l) {
    hipMemsetAsync(agg, 0, (size_t)N * H * sizeof(float), stream);
    gnn_edge<<<nblkE, 256, 0, stream>>>(hc, erow, ecol, edges_features,
                                        We1F[l], be1 + l * 64, We2F[l], be2 + l * 64,
                                        agg, E);
    gnn_node<<<nblkNode, 256, 0, stream>>>(hc, agg,
                                           Wn1F[l], bn1 + l * 64, Wn2F[l], bn2 + l * 64,
                                           hn, N);
    float* t = hc; hc = hn; hn = t;
  }
  gnn_unembed<<<nblkN, 256, 0, stream>>>(hc, Wu1F, bu1, Wu2F, bu2, (float*)d_out, N);
}
